// SpectralDynamicGraphBuilder_85864986182002
// MI455X (gfx1250) — matrix operand order repaired and checked by reading, not compiled, not run on hardware
//
#include <hip/hip_runtime.h>
#include <math.h>

typedef __attribute__((ext_vector_type(16))) _Float16 v16h;
typedef __attribute__((ext_vector_type(8)))  _Float16 v8h;
typedef __attribute__((ext_vector_type(8)))  float    v8f;

#define Bc 2
#define Tc 128
#define Nn 4096
#define Fc 32
#define Kk 10
#define NB 64
#define LSEG 64
#define NFR 33
#define NSEG 3
// log2-domain temperature scale: sim/TEMP * log2(e)
#define INV_TEMP_LOG2E (1.4426950408889634f / 0.07f)
#define NEG_BIG (-3.0e38f)

// ---------------------------------------------------------------------------
// Kernel 0: softmax of the 32 feature logits (tiny, one thread).
// ---------------------------------------------------------------------------
__global__ void featw_kernel(const float* __restrict__ logits,
                             float* __restrict__ featw) {
  if (threadIdx.x == 0 && blockIdx.x == 0) {
    float m = NEG_BIG;
    for (int i = 0; i < Fc; ++i) m = fmaxf(m, logits[i]);
    float e[Fc];
    float s = 0.f;
    for (int i = 0; i < Fc; ++i) { e[i] = expf(logits[i] - m); s += e[i]; }
    float inv = 1.0f / s;
    for (int i = 0; i < Fc; ++i) featw[i] = e[i] * inv;
  }
}

// ---------------------------------------------------------------------------
// Kernel 1: per (b,n) node -> Welch PSD -> weighted log-power -> band proj
//           -> LayerNorm -> L2 normalize -> f16 feature row fn[b*N+n][64].
// One workgroup (256 threads) per node; x tile staged in LDS.
// ---------------------------------------------------------------------------
__global__ __launch_bounds__(256) void band_feat_kernel(
    const float* __restrict__ x, const float* __restrict__ featw,
    const float* __restrict__ bandW, _Float16* __restrict__ fnH) {
  __shared__ float X[Tc * Fc];   // 16KB: X[t*32 + i]
  __shared__ float win[LSEG];
  __shared__ float ctab[LSEG];
  __shared__ float stab[LSEG];
  __shared__ float pagg[NFR];
  __shared__ float logp[NFR];
  __shared__ float feat[NB];
  __shared__ float red[4];

  const int tid = threadIdx.x;
  const int bn  = blockIdx.x;        // b*4096 + n
  const int n   = bn & (Nn - 1);
  const int b   = bn >> 12;

  // Stage x[b, :, n, :] -> LDS (128 rows of 32 contiguous floats)
  const float* xp = x + ((size_t)b * Tc) * ((size_t)Nn * Fc) + (size_t)n * Fc;
  for (int e = tid; e < Tc * Fc; e += 256) {
    int t = e >> 5, i = e & 31;
    X[e] = xp[(size_t)t * (Nn * Fc) + i];
  }
  if (tid < LSEG) {
    float ph = 2.0f * 3.14159265358979f * (float)tid / (float)LSEG;
    win[tid]  = 0.5f * (1.0f - cosf(ph));   // periodic Hann
    ctab[tid] = cosf(ph);
    stab[tid] = sinf(ph);
  }
  if (tid < NFR) pagg[tid] = 0.0f;
  __syncthreads();

  // DFT power per (freq, feature); 33*32 = 1056 tasks over 256 threads.
  for (int task = tid; task < NFR * Fc; task += 256) {
    int fr = task >> 5, i = task & 31;
    float acc = 0.f;
    for (int s = 0; s < NSEG; ++s) {
      int st = s * 32;                       // starts = 0, 32, 64
      float re = 0.f, im = 0.f;
      for (int t = 0; t < LSEG; ++t) {
        float v = win[t] * X[(st + t) * Fc + i];
        int k = (fr * t) & 63;
        re += v * ctab[k];
        im += v * stab[k];
      }
      acc += re * re + im * im;
    }
    atomicAdd(&pagg[fr], acc * (1.0f / NSEG) * featw[i]);
  }
  __syncthreads();

  if (tid < NFR) logp[tid] = logf(fmaxf(pagg[tid], 1e-8f));
  __syncthreads();

  if (tid < NB) {
    const float* wrow = bandW + tid * NFR;
    float a = 0.f;
    for (int fr = 0; fr < NFR; ++fr) a += logp[fr] * wrow[fr];
    feat[tid] = a;
  }
  __syncthreads();

  if (tid == 0) {                            // LayerNorm stats over 64 bands
    float mu = 0.f;
    for (int j = 0; j < NB; ++j) mu += feat[j];
    mu *= (1.0f / NB);
    float var = 0.f;
    for (int j = 0; j < NB; ++j) { float d = feat[j] - mu; var += d * d; }
    var *= (1.0f / NB);
    red[0] = mu;
    red[1] = rsqrtf(var + 1e-5f);
  }
  __syncthreads();
  if (tid < NB) feat[tid] = (feat[tid] - red[0]) * red[1];
  __syncthreads();
  if (tid == 0) {                            // L2 norm for cosine
    float ss = 0.f;
    for (int j = 0; j < NB; ++j) ss += feat[j] * feat[j];
    red[2] = 1.0f / fmaxf(sqrtf(ss), 1e-8f);
  }
  __syncthreads();
  if (tid < NB) fnH[(size_t)bn * NB + tid] = (_Float16)(feat[tid] * red[2]);
}

// ---------------------------------------------------------------------------
// Kernel 2: fused cosine-sim GEMM (WMMA f16->f32) + tempered softmax denom
//           + per-row top-10 + symmetric sparse scatter.
// One WG (8 waves) handles 16 rows of one batch; columns in 2 halves of 2048.
// All exp work is done in the log2 domain so the HW v_exp_f32 can be used
// (arguments are bounded by cos-sim: |v| <= log2e/TEMP ~ 20.6).
// The diagonal is included in the softmax denominator but excluded from
// top-k by a one-shot fixup of the 16 diagonal slots (not a per-element
// compare in the hot loop).
// ---------------------------------------------------------------------------
__global__ __launch_bounds__(256) void sim_topk_kernel(
    const _Float16* __restrict__ fnH, float* __restrict__ out) {
  __shared__ float rowBuf[16][2048];         // 128KB half-row staging
  __shared__ float expSum[16];
  __shared__ float candV[16][2 * Kk];
  __shared__ int   candI[16][2 * Kk];

  const int tid  = threadIdx.x;
  const int lane = tid & 31;
  const int wave = tid >> 5;                 // 0..7
  const int wg   = blockIdx.x;               // B * (N/16)
  const int b    = wg >> 8;
  const int r0   = (wg & 255) << 4;          // row tile base
  const _Float16* fb = fnH + (size_t)b * Nn * NB;

  if (tid < 16) expSum[tid] = 0.f;

  // --- A fragments: rows r0..r0+15, K=64 as two 16x16x32 chunks.
  // ISA 16-bit A layout: lanes 0-15 -> K 0..7 & 16..23; lanes 16-31 -> +8.
  v16h aF[2];
  {
    const int m = lane & 15;
    const int o = (lane < 16) ? 0 : 8;
    const _Float16* arow = fb + (size_t)(r0 + m) * NB;
#pragma unroll
    for (int kc = 0; kc < 2; ++kc) {
      v8h lo = *(const v8h*)(arow + kc * 32 + o);
      v8h hi = *(const v8h*)(arow + kc * 32 + o + 16);
      v16h a;
#pragma unroll
      for (int j = 0; j < 8; ++j) { a[j] = lo[j]; a[j + 8] = hi[j]; }
      aF[kc] = a;
    }
  }

  float part[8];                             // exp2 partial sums per D-row reg
#pragma unroll
  for (int r = 0; r < 8; ++r) part[r] = 0.f;
  const int rbase = (lane >> 4) << 3;        // D rows: lanes>=16 hold M=r+8
  const int cl    = lane & 15;               // D col within tile

  for (int half = 0; half < 2; ++half) {
    // Each wave computes 16 column tiles of its stripe.
#pragma unroll 2
    for (int tt = 0; tt < 16; ++tt) {
      const int c0 = half * 2048 + (wave * 16 + tt) * 16;
      v8f acc = {};
#pragma unroll
      for (int kc = 0; kc < 2; ++kc) {
        // B layout: lane = column, K 0..7 & 16..23 (+8 for lanes>=16), same as A
        const _Float16* bcol =
            fb + (size_t)(c0 + cl) * NB + kc * 32 + ((lane < 16) ? 0 : 8);
        v8h blo = *(const v8h*)bcol;
        v8h bhi = *(const v8h*)(bcol + 16);
        v16h bF;
#pragma unroll
        for (int j = 0; j < 8; ++j) { bF[j] = blo[j]; bF[j + 8] = bhi[j]; }
        acc = __builtin_amdgcn_wmma_f32_16x16x32_f16(
            false, aF[kc], false, bF, (short)0, acc, false, false);
      }
      const int colL = (c0 & 2047) + cl;     // column within this half
#pragma unroll
      for (int r = 0; r < 8; ++r) {
        float v = acc[r] * INV_TEMP_LOG2E;           // log2-domain logit
        part[r] += __builtin_amdgcn_exp2f(v);        // HW v_exp_f32; diag in
        rowBuf[rbase + r][colL] = v;                 // unconditional store
      }
    }
    __syncthreads();

    // One-shot diagonal knockout: the 16 diagonal columns [r0, r0+16) lie
    // entirely inside one half (r0 is 16-aligned, halves are 2048-aligned).
    if ((r0 >> 11) == half && tid < 16) {
      rowBuf[tid][(r0 & 2047) + tid] = NEG_BIG;
    }
    __syncthreads();

    // Top-10 of this half per row; wave w owns rows 2w, 2w+1 (wave-private).
    for (int k = 0; k < Kk; ++k) {
#pragma unroll
      for (int rr = 0; rr < 2; ++rr) {
        const int row = wave * 2 + rr;
        float bv = NEG_BIG;
        int   bi = 0;
        for (int j = lane; j < 2048; j += 32) {
          float v = rowBuf[row][j];
          if (v > bv) { bv = v; bi = j; }
        }
        for (int off = 16; off > 0; off >>= 1) {   // wave32 argmax butterfly
          float ov = __shfl_xor(bv, off, 32);
          int   oi = __shfl_xor(bi, off, 32);
          if (ov > bv || (ov == bv && oi < bi)) { bv = ov; bi = oi; }
        }
        if (lane == 0) {
          candV[row][half * Kk + k] = bv;
          candI[row][half * Kk + k] = half * 2048 + bi;
          rowBuf[row][bi] = NEG_BIG;               // mark taken
        }
      }
    }
    __syncthreads();                               // before rowBuf reuse
  }

  // Softmax denominator per row.
#pragma unroll
  for (int r = 0; r < 8; ++r) atomicAdd(&expSum[rbase + r], part[r]);
  __syncthreads();

  // Merge 20 candidates -> top 10; symmetric sparse scatter of 0.5*softmax.
  if (tid < 16) {
    const int   row  = tid;
    const int   rowg = r0 + row;
    const float inv  = 0.5f / expSum[row];
    bool used[2 * Kk];
#pragma unroll
    for (int j = 0; j < 2 * Kk; ++j) used[j] = false;
    for (int k = 0; k < Kk; ++k) {
      float bv = NEG_BIG;
      int   bj = 0;
      for (int j = 0; j < 2 * Kk; ++j)
        if (!used[j] && candV[row][j] > bv) { bv = candV[row][j]; bj = j; }
      used[bj] = true;
      const int   col = candI[row][bj];
      const float p   = __builtin_amdgcn_exp2f(bv) * inv;
      float* base = out + (size_t)b * Nn * Nn;
      atomicAdd(base + (size_t)rowg * Nn + col, p);
      atomicAdd(base + (size_t)col * Nn + rowg, p);
    }
  }
}

// ---------------------------------------------------------------------------
extern "C" void kernel_launch(void* const* d_in, const int* in_sizes, int n_in,
                              void* d_out, int out_size, void* d_ws,
                              size_t ws_size, hipStream_t stream) {
  const float* x      = (const float*)d_in[0];  // [B,T,N,F]
  const float* logits = (const float*)d_in[1];  // [F]
  const float* bandW  = (const float*)d_in[2];  // [64,33]
  float* out = (float*)d_out;                   // [B,N,N]

  float*    featw = (float*)d_ws;                          // 32 floats
  _Float16* fnH   = (_Float16*)((char*)d_ws + 256);        // [B*N, 64] f16

  hipMemsetAsync(d_out, 0, (size_t)out_size * sizeof(float), stream);
  featw_kernel<<<1, 32, 0, stream>>>(logits, featw);
  band_feat_kernel<<<Bc * Nn, 256, 0, stream>>>(x, featw, bandW, fnH);
  sim_topk_kernel<<<Bc * (Nn / 16), 256, 0, stream>>>(fnH, out);
}
